// PoolingLayer_29678224015886
// MI455X (gfx1250) — compile-verified, matrix operand order checked by reading, not run on hardware
//
#include <hip/hip_runtime.h>
#include <hip/hip_bf16.h>

// ---------------------------------------------------------------------------
// Fused "pooling transformer cell" for MI455X (gfx1250, wave32, WMMA).
//   y = INorm( s + FFN(s) ),  s = INorm( parent + OutProj(Attn(QKV(ctx))) )
//
// Roofline: 1.24 TFLOP total vs 0.8 GB mandatory HBM -> HBM-bound at ~35us
// only if L2 weight traffic is tamed. This version: BM=32 rows/block, every
// B-panel (1 KB) load feeds 2-6 WMMAs (M/N blocking), so L2 read volume is
// 2048 blocks x 13.5 MB = 27 GB and per-panel intensity is 32-96 FLOP/B.
// Q is never materialized: the Q GEMM is fused with logit evaluation via
// cross-lane shfl reductions + ds_add_f32 into a shared logits buffer.
// ---------------------------------------------------------------------------

typedef __attribute__((ext_vector_type(16))) __bf16 v16bf;
typedef __attribute__((ext_vector_type(8)))  float  v8f;

static constexpr int DMODEL = 768;
static constexpr int DINNER = 3072;
static constexpr int NHEAD  = 12;
static constexpr int BM     = 32;            // batch rows per block (2 WMMA M-tiles)
static constexpr int KT_D   = DMODEL / 32;   // 24 k-steps (K=768)
static constexpr int KT_DI  = DINNER / 32;   // 96 k-steps (K=3072)

// packed-weight offsets in bf16 elements inside d_ws
static constexpr size_t WQ_OFF = 0;
static constexpr size_t WK_OFF = (size_t)768 * 768;
static constexpr size_t WV_OFF = (size_t)2 * 768 * 768;
static constexpr size_t OW_OFF = (size_t)3 * 768 * 768;
static constexpr size_t W1_OFF = (size_t)4 * 768 * 768;
static constexpr size_t W2_OFF = (size_t)4 * 768 * 768 + (size_t)3072 * 768;

// LDS overlay map (bytes); peak live set 292.5 KB < 320 KB WGP limit.
static constexpr int X_OFF    = 0;        // x_kv : 96x768  bf16 = 147456
static constexpr int K_OFF    = 147456;   // k    : 96x768  bf16 = 147456
static constexpr int ATTN_OFF = 294912;   // attn : 32x12x3 f32  = 4608
static constexpr int O_OFF    = 147456;   // o    : 32x768  bf16 (over k head)
static constexpr int SRCB_OFF = 196608;   // src  : 32x768  bf16 (over k mid)
static constexpr int H_OFF    = 0;        // h    : 32x3072 bf16 (over x_kv + o)
static constexpr int YPRE_OFF = 245760;   // ypre : 32x768  bf16 (over k tail)
static constexpr int SMEM_BYTES = 299520;

__device__ __forceinline__ unsigned short f2bf(float f) {
    unsigned int u = __float_as_uint(f);
    u += 0x7fffu + ((u >> 16) & 1u);        // round-to-nearest-even
    return (unsigned short)(u >> 16);
}
__device__ __forceinline__ float bf2f(unsigned short b) {
    return __uint_as_float(((unsigned int)b) << 16);
}

union BFV { uint4 u[2]; v16bf v; };

// A operand: LDS row-major bf16, WMMA 16x32 bf16 A lane layout (2x ds_load_b128).
__device__ __forceinline__ v16bf load_a(const unsigned short* base, int ldk,
                                        int row, int kt, int lane) {
    const unsigned short* p = base + row * ldk + (kt << 5) + ((lane >> 4) << 3);
    BFV t;
    t.u[0] = *(const uint4*)(p);
    t.u[1] = *(const uint4*)(p + 16);
    return t.v;
}

// B operand: pre-swizzled panel, 32 contiguous bytes/lane (2x global_load_b128).
__device__ __forceinline__ v16bf load_b(const unsigned short* w, int tile, int lane) {
    const unsigned short* p = w + ((size_t)tile << 9) + (lane << 4);
    BFV t;
    t.u[0] = *(const uint4*)(p);
    t.u[1] = *(const uint4*)(p + 8);
    return t.v;
}

__device__ __forceinline__ v8f wmma_bf16(v16bf a, v16bf b, v8f c) {
    return __builtin_amdgcn_wmma_f32_16x16x32_bf16(false, a, false, b,
                                                   (short)0, c, false, false);
}

// ---------------------------------------------------------------------------
// Weight pre-pack: f32 (N,K) row-major -> bf16 WMMA-B panels of 512 elems.
// ---------------------------------------------------------------------------
__global__ void pack_weights(const float* __restrict__ src,
                             unsigned short* __restrict__ dst,
                             int N, int K, int total) {
    int e = blockIdx.x * 256 + threadIdx.x;
    if (e >= total) return;
    int KT   = K >> 5;
    int tile = e >> 9;
    int r    = e & 511;
    int l    = r >> 4;
    int j    = r & 15;
    int nt   = tile / KT;
    int kt   = tile - nt * KT;
    int n    = (nt << 4) + (l & 15);
    int k    = (kt << 5) + ((l >> 4) << 3) + (j & 7) + ((j >= 8) ? 16 : 0);
    dst[e]   = f2bf(src[(size_t)n * K + k]);
}

// ---------------------------------------------------------------------------
// Fused main kernel: 256 threads (8 wave32), BM=32 batch rows per block.
// ---------------------------------------------------------------------------
__global__ __launch_bounds__(256) void fused_cell(
    const float* __restrict__ children,   // (B,2,768)
    const float* __restrict__ parent,     // (B,768)
    const float* __restrict__ pos_emb,    // (3,768)
    const float* __restrict__ in_proj_b,  // (2304)
    const float* __restrict__ out_b,      // (768)
    const float* __restrict__ b1,         // (3072)
    const float* __restrict__ b2,         // (768)
    const unsigned short* __restrict__ wpack,
    float* __restrict__ out)              // (B,768)
{
    __shared__ __align__(16) unsigned char smem[SMEM_BYTES];
    unsigned short* x_kv = (unsigned short*)(smem + X_OFF);    // 96 flat kv rows
    unsigned short* kbuf = (unsigned short*)(smem + K_OFF);    // 96 K rows
    float*          attn = (float*)(smem + ATTN_OFF);          // 32x12x3 logits
    unsigned short* obuf = (unsigned short*)(smem + O_OFF);    // 32x768 attn out
    unsigned short* srcb = (unsigned short*)(smem + SRCB_OFF); // 32x768 src (bf16)
    unsigned short* hbuf = (unsigned short*)(smem + H_OFF);    // 32x3072 gelu out
    unsigned short* ypre = (unsigned short*)(smem + YPRE_OFF); // 32x768 pre-norm2

    const int tid   = threadIdx.x;
    const int wave  = tid >> 5;
    const int lane  = tid & 31;
    const int item0 = blockIdx.x * BM;

    // ---- Phase A: stage kv_in = [parent;cl;cr] + pos_emb as bf16; zero logits
    for (int i = 0; i < 72; ++i) {
        int g   = tid + (i << 8);               // 0..18431 float4 groups
        int row = g / 192;                      // flat kv row = item*3 + j
        int c   = (g - row * 192) << 2;
        int j   = row % 3;
        int it  = row / 3;
        const float* s = (j == 0)
            ? (parent + (size_t)(item0 + it) * DMODEL + c)
            : (children + ((size_t)(item0 + it) * 2 + (j - 1)) * DMODEL + c);
        float4 a = *(const float4*)s;
        float4 p = *(const float4*)(pos_emb + j * DMODEL + c);
        ushort4 o4;
        o4.x = f2bf(a.x + p.x); o4.y = f2bf(a.y + p.y);
        o4.z = f2bf(a.z + p.z); o4.w = f2bf(a.w + p.w);
        *(ushort4*)(x_kv + row * DMODEL + c) = o4;
    }
    for (int t = tid; t < BM * NHEAD * 3; t += 256) attn[t] = 0.f;
    __syncthreads();

    // ---- Phase B: K projection. One B-panel load feeds 6 WMMAs (6 M-tiles).
    for (int s = 0; s < 6; ++s) {
        int nt = wave + (s << 3);
        v8f acc[6] = {};
        int tile0 = nt * KT_D;
        __builtin_prefetch(wpack + WK_OFF + ((size_t)tile0 << 9), 0, 1);
#pragma unroll 2
        for (int kt = 0; kt < KT_D; ++kt) {
            v16bf b = load_b(wpack + WK_OFF, tile0 + kt, lane);
#pragma unroll
            for (int mt = 0; mt < 6; ++mt) {
                v16bf a = load_a(x_kv, DMODEL, mt * 16 + (lane & 15), kt, lane);
                acc[mt] = wmma_bf16(a, b, acc[mt]);
            }
        }
        int n = (nt << 4) + (lane & 15);
        float bk = in_proj_b[DMODEL + n];
#pragma unroll
        for (int mt = 0; mt < 6; ++mt)
#pragma unroll
            for (int r = 0; r < 8; ++r) {
                int row = mt * 16 + r + ((lane >> 4) << 3);
                kbuf[row * DMODEL + n] = f2bf(acc[mt][r] + bk);
            }
    }
    __syncthreads();

    // ---- Phase C: Q projection fused with logits (Q never materialized).
    // Each lane holds q[item, n]; dot against K in LDS, reduce over the 16
    // lanes that share an item (shfl_xor), accumulate via ds_add_f32.
    for (int s = 0; s < 6; ++s) {
        int nt = wave + (s << 3);
        int head = nt >> 2;                     // 4 n-tiles per head
        v8f acc[2] = {};
        int tile0 = nt * KT_D;
#pragma unroll 4
        for (int kt = 0; kt < KT_D; ++kt) {
            v16bf b = load_b(wpack + WQ_OFF, tile0 + kt, lane);
#pragma unroll
            for (int mt = 0; mt < 2; ++mt) {
                v16bf a = load_a(x_kv, DMODEL, (mt * 16 + (lane & 15)) * 3, kt, lane);
                acc[mt] = wmma_bf16(a, b, acc[mt]);
            }
        }
        int n = (nt << 4) + (lane & 15);
        float bq = in_proj_b[n];
#pragma unroll
        for (int mt = 0; mt < 2; ++mt)
#pragma unroll
            for (int r = 0; r < 8; ++r) {
                int item = mt * 16 + r + ((lane >> 4) << 3);
                float qv = acc[mt][r] + bq;
                for (int j = 0; j < 3; ++j) {
                    float val = qv * bf2f(kbuf[(item * 3 + j) * DMODEL + n]);
                    val += __shfl_xor(val, 1, 32);
                    val += __shfl_xor(val, 2, 32);
                    val += __shfl_xor(val, 4, 32);
                    val += __shfl_xor(val, 8, 32);
                    if ((lane & 15) == 0)
                        atomicAdd(&attn[(item * NHEAD + head) * 3 + j], val);
                }
            }
    }
    __syncthreads();

    // ---- Phase D: softmax over the 3 keys (scale 1/sqrt(64))
    for (int t = tid; t < BM * NHEAD; t += 256) {
        float l0 = attn[t * 3 + 0] * 0.125f;
        float l1 = attn[t * 3 + 1] * 0.125f;
        float l2 = attn[t * 3 + 2] * 0.125f;
        float mx = fmaxf(l0, fmaxf(l1, l2));
        float e0 = __expf(l0 - mx), e1 = __expf(l1 - mx), e2 = __expf(l2 - mx);
        float inv = 1.f / (e0 + e1 + e2);
        attn[t * 3 + 0] = e0 * inv;
        attn[t * 3 + 1] = e1 * inv;
        attn[t * 3 + 2] = e2 * inv;
    }
    __syncthreads();

    // ---- Phase E: V projection fused with attention combine.
    // One B-panel load feeds 6 WMMAs (2 item-tiles x 3 ctx rows).
    for (int s = 0; s < 6; ++s) {
        int nt = wave + (s << 3);
        int head = nt >> 2;
        v8f acc[6] = {};
        int tile0 = nt * KT_D;
#pragma unroll 2
        for (int kt = 0; kt < KT_D; ++kt) {
            v16bf b = load_b(wpack + WV_OFF, tile0 + kt, lane);
#pragma unroll
            for (int mt = 0; mt < 2; ++mt)
#pragma unroll
                for (int j = 0; j < 3; ++j) {
                    v16bf a = load_a(x_kv, DMODEL,
                                     (mt * 16 + (lane & 15)) * 3 + j, kt, lane);
                    acc[mt * 3 + j] = wmma_bf16(a, b, acc[mt * 3 + j]);
                }
        }
        int n = (nt << 4) + (lane & 15);
        float bv = in_proj_b[2 * DMODEL + n];
#pragma unroll
        for (int mt = 0; mt < 2; ++mt)
#pragma unroll
            for (int r = 0; r < 8; ++r) {
                int item = mt * 16 + r + ((lane >> 4) << 3);
                const float* ap = attn + (item * NHEAD + head) * 3;
                float o = ap[0] * (acc[mt * 3 + 0][r] + bv)
                        + ap[1] * (acc[mt * 3 + 1][r] + bv)
                        + ap[2] * (acc[mt * 3 + 2][r] + bv);
                obuf[item * DMODEL + n] = f2bf(o);
            }
    }
    __syncthreads();

    // ---- Phase F: out projection (MT2 x NT2) + parent residual -> srcb
    for (int s = 0; s < 3; ++s) {
        int nt0 = (wave + (s << 3)) << 1;
        v8f acc[4] = {};
#pragma unroll 4
        for (int kt = 0; kt < KT_D; ++kt) {
            v16bf b0 = load_b(wpack + OW_OFF, nt0 * KT_D + kt, lane);
            v16bf b1 = load_b(wpack + OW_OFF, (nt0 + 1) * KT_D + kt, lane);
#pragma unroll
            for (int mt = 0; mt < 2; ++mt) {
                v16bf a = load_a(obuf, DMODEL, mt * 16 + (lane & 15), kt, lane);
                acc[mt * 2 + 0] = wmma_bf16(a, b0, acc[mt * 2 + 0]);
                acc[mt * 2 + 1] = wmma_bf16(a, b1, acc[mt * 2 + 1]);
            }
        }
#pragma unroll
        for (int q = 0; q < 2; ++q) {
            int n = ((nt0 + q) << 4) + (lane & 15);
            float bo = out_b[n];
#pragma unroll
            for (int mt = 0; mt < 2; ++mt)
#pragma unroll
                for (int r = 0; r < 8; ++r) {
                    int item = mt * 16 + r + ((lane >> 4) << 3);
                    float pv = parent[(size_t)(item0 + item) * DMODEL + n];
                    srcb[item * DMODEL + n] = f2bf(acc[mt * 2 + q][r] + bo + pv);
                }
        }
    }
    __syncthreads();

    // ---- Phase G: InstanceNorm #1 in-place on srcb (4 rows per wave)
    for (int rr = 0; rr < 4; ++rr) {
        int row = (wave << 2) + rr;
        float sum = 0.f, sq = 0.f;
        for (int i = 0; i < 24; ++i) {
            float v = bf2f(srcb[row * DMODEL + lane + (i << 5)]);
            sum += v; sq += v * v;
        }
        for (int off = 16; off; off >>= 1) {
            sum += __shfl_xor(sum, off, 32);
            sq  += __shfl_xor(sq,  off, 32);
        }
        float mean = sum * (1.f / 768.f);
        float var  = sq * (1.f / 768.f) - mean * mean;
        float rstd = rsqrtf(var + 1e-5f);
        for (int i = 0; i < 24; ++i) {
            int c = lane + (i << 5);
            float v = (bf2f(srcb[row * DMODEL + c]) - mean) * rstd;
            srcb[row * DMODEL + c] = f2bf(v);
        }
    }
    __syncthreads();

    // ---- Phase H: FFN1 (768 -> 3072, MT2 x NT2) + exact GELU -> hbuf
    for (int s = 0; s < 12; ++s) {
        int nt0 = (wave + (s << 3)) << 1;       // 0..190
        v8f acc[4] = {};
#pragma unroll 4
        for (int kt = 0; kt < KT_D; ++kt) {
            v16bf b0 = load_b(wpack + W1_OFF, nt0 * KT_D + kt, lane);
            v16bf b1 = load_b(wpack + W1_OFF, (nt0 + 1) * KT_D + kt, lane);
#pragma unroll
            for (int mt = 0; mt < 2; ++mt) {
                v16bf a = load_a(srcb, DMODEL, mt * 16 + (lane & 15), kt, lane);
                acc[mt * 2 + 0] = wmma_bf16(a, b0, acc[mt * 2 + 0]);
                acc[mt * 2 + 1] = wmma_bf16(a, b1, acc[mt * 2 + 1]);
            }
        }
#pragma unroll
        for (int q = 0; q < 2; ++q) {
            int n = ((nt0 + q) << 4) + (lane & 15);
            float bb = b1[n];
#pragma unroll
            for (int mt = 0; mt < 2; ++mt)
#pragma unroll
                for (int r = 0; r < 8; ++r) {
                    int item = mt * 16 + r + ((lane >> 4) << 3);
                    float x = acc[mt * 2 + q][r] + bb;
                    float g = 0.5f * x * (1.f + erff(x * 0.70710678f));
                    hbuf[item * DINNER + n] = f2bf(g);
                }
        }
    }
    __syncthreads();

    // ---- Phase I: FFN2 (3072 -> 768, MT2 x NT2) + residual -> ypre
    for (int s = 0; s < 3; ++s) {
        int nt0 = (wave + (s << 3)) << 1;
        v8f acc[4] = {};
#pragma unroll 2
        for (int kt = 0; kt < KT_DI; ++kt) {
            v16bf b0 = load_b(wpack + W2_OFF, nt0 * KT_DI + kt, lane);
            v16bf b1 = load_b(wpack + W2_OFF, (nt0 + 1) * KT_DI + kt, lane);
#pragma unroll
            for (int mt = 0; mt < 2; ++mt) {
                v16bf a = load_a(hbuf, DINNER, mt * 16 + (lane & 15), kt, lane);
                acc[mt * 2 + 0] = wmma_bf16(a, b0, acc[mt * 2 + 0]);
                acc[mt * 2 + 1] = wmma_bf16(a, b1, acc[mt * 2 + 1]);
            }
        }
#pragma unroll
        for (int q = 0; q < 2; ++q) {
            int n = ((nt0 + q) << 4) + (lane & 15);
            float bb = b2[n];
#pragma unroll
            for (int mt = 0; mt < 2; ++mt)
#pragma unroll
                for (int r = 0; r < 8; ++r) {
                    int item = mt * 16 + r + ((lane >> 4) << 3);
                    float v = acc[mt * 2 + q][r] + bb
                            + bf2f(srcb[item * DMODEL + n]);
                    ypre[item * DMODEL + n] = f2bf(v);
                }
        }
    }
    __syncthreads();

    // ---- Phase J: InstanceNorm #2 -> global output (coalesced f32 stores)
    for (int rr = 0; rr < 4; ++rr) {
        int row = (wave << 2) + rr;
        float sum = 0.f, sq = 0.f;
        for (int i = 0; i < 24; ++i) {
            float v = bf2f(ypre[row * DMODEL + lane + (i << 5)]);
            sum += v; sq += v * v;
        }
        for (int off = 16; off; off >>= 1) {
            sum += __shfl_xor(sum, off, 32);
            sq  += __shfl_xor(sq,  off, 32);
        }
        float mean = sum * (1.f / 768.f);
        float var  = sq * (1.f / 768.f) - mean * mean;
        float rstd = rsqrtf(var + 1e-5f);
        for (int i = 0; i < 24; ++i) {
            int c = lane + (i << 5);
            out[(size_t)(item0 + row) * DMODEL + c] =
                (bf2f(ypre[row * DMODEL + c]) - mean) * rstd;
        }
    }
}

extern "C" void kernel_launch(void* const* d_in, const int* in_sizes, int n_in,
                              void* d_out, int out_size, void* d_ws, size_t ws_size,
                              hipStream_t stream) {
    const float* children  = (const float*)d_in[0];
    const float* parent    = (const float*)d_in[1];
    const float* pos_emb   = (const float*)d_in[2];
    const float* in_proj_w = (const float*)d_in[3];
    const float* in_proj_b = (const float*)d_in[4];
    const float* out_w     = (const float*)d_in[5];
    const float* out_b     = (const float*)d_in[6];
    const float* w1        = (const float*)d_in[7];
    const float* b1        = (const float*)d_in[8];
    const float* w2        = (const float*)d_in[9];
    const float* b2        = (const float*)d_in[10];
    float* out             = (float*)d_out;
    unsigned short* wp     = (unsigned short*)d_ws;   // 13.5 MB bf16 panels

    const int B = in_sizes[1] / DMODEL;               // 65536

    auto pack = [&](const float* src, size_t off, int N, int K) {
        int total = N * K;
        pack_weights<<<(total + 255) / 256, 256, 0, stream>>>(src, wp + off, N, K, total);
    };
    pack(in_proj_w,                 WQ_OFF, 768, 768);
    pack(in_proj_w + 768 * 768,     WK_OFF, 768, 768);
    pack(in_proj_w + 2 * 768 * 768, WV_OFF, 768, 768);
    pack(out_w,                     OW_OFF, 768, 768);
    pack(w1,                        W1_OFF, 3072, 768);
    pack(w2,                        W2_OFF, 768, 3072);

    fused_cell<<<B / BM, 256, 0, stream>>>(children, parent, pos_emb, in_proj_b,
                                           out_b, b1, b2, wp, out);
}
